// VisionRetentionAll_9182640078865
// MI455X (gfx1250) — matrix operand order repaired and checked by reading, compiled, not run on hardware
//
#include <hip/hip_runtime.h>

#define DIM_B  8
#define DIM_H  32
#define DIM_W  32
#define DIM_C  768
#define DIM_NH 12
#define DIM_HD 64
#define DIM_L  1024   // H*W
#define DIM_M  8192   // B*L

typedef __attribute__((ext_vector_type(16))) __bf16 v16bf;
typedef __attribute__((ext_vector_type(8)))  float  v8f;

static __device__ __forceinline__ v8f v8f_zero() {
  v8f r;
#pragma unroll
  for (int i = 0; i < 8; ++i) r[i] = 0.0f;
  return r;
}

// --- CDNA5 async global->LDS copy (16 bytes per lane), ASYNCcnt-tracked -----
static __device__ __forceinline__ void async_copy_b128(const __bf16* g, __bf16* l) {
  unsigned lds_addr = (unsigned)(size_t)(void*)l;            // LDS aperture low 32b
  unsigned long long gaddr = (unsigned long long)(size_t)(const void*)g;
  asm volatile("global_load_async_to_lds_b128 %0, %1, off"
               :: "v"(lds_addr), "v"(gaddr)
               : "memory");
}

#if __has_builtin(__builtin_amdgcn_s_wait_asynccnt)
#define WAIT_ASYNC(n) __builtin_amdgcn_s_wait_asynccnt(n)
#else
#define WAIT_ASYNC(n) asm volatile("s_wait_asynccnt %0" :: "i"(n) : "memory")
#endif

// ---------------------------------------------------------------- f32 -> bf16
__global__ __launch_bounds__(256) void k_cvt_bf16(const float* __restrict__ s,
                                                  __bf16* __restrict__ d, int n) {
  int i = (blockIdx.x * 256 + threadIdx.x) * 4;
  if (i + 3 < n) {
    float4 v = *(const float4*)(s + i);
    d[i + 0] = (__bf16)v.x;
    d[i + 1] = (__bf16)v.y;
    d[i + 2] = (__bf16)v.z;
    d[i + 3] = (__bf16)v.w;
  }
}

#define LDA_PAD 40

// ------------------------------------------- fused QKV GEMM (shared A tiles)
// out_i[m,n] = (sum_k X[m,k]*W_i[n,k] + b_i[n]) * s_i  for i in {q,k,v}
// Block: 256 threads (8 waves), tile 128(M) x 64(N), K panels of 32,
// double-buffered async global->LDS staging + ping-pong B fragments.
__global__ __launch_bounds__(256) void k_gemm_qkv(const __bf16* __restrict__ X,
    const __bf16* __restrict__ W0, const __bf16* __restrict__ W1,
    const __bf16* __restrict__ W2,
    const float* __restrict__ b0, const float* __restrict__ b1,
    const float* __restrict__ b2,
    float* __restrict__ o0, float* __restrict__ o1, float* __restrict__ o2,
    float s1scale)
{
  __shared__ __bf16 As[2][128 * LDA_PAD];
  __shared__ __bf16 Bs[3][2][64 * LDA_PAD];
  const int tid  = threadIdx.x;
  const int wave = tid >> 5, lane = tid & 31;
  const int hb = lane >> 4, lm = lane & 15;
  const int m0 = blockIdx.y * 128, n0 = blockIdx.x * 64;
  const int K = DIM_C, N = DIM_C;

  v8f acc[3][4];
#pragma unroll
  for (int g = 0; g < 3; ++g)
#pragma unroll
    for (int t = 0; t < 4; ++t) acc[g][t] = v8f_zero();

  const int arow = tid >> 1, ach = tid & 1;   // 128 rows x 2 chunks of 16
  const int brow = tid >> 2, bch = tid & 3;   // 64 rows x 4 chunks of 8

  const __bf16* Wg[3] = {W0, W1, W2};

  // issue one K-panel of async loads into buffer p (5 loads/thread)
  auto issue = [&](int kk, int p) {
    const __bf16* asrc = X + (size_t)(m0 + arow) * K + kk + ach * 16;
    __bf16*       adst = As[p] + arow * LDA_PAD + ach * 16;
    async_copy_b128(asrc,     adst);
    async_copy_b128(asrc + 8, adst + 8);
#pragma unroll
    for (int g = 0; g < 3; ++g)
      async_copy_b128(Wg[g] + (size_t)(n0 + brow) * K + kk + bch * 8,
                      Bs[g][p] + brow * LDA_PAD + bch * 8);
  };

  issue(0, 0);
  for (int kk = 0, it = 0; kk < K; kk += 32, ++it) {
    const int p = it & 1;
    const bool more = (kk + 32 < K);
    if (more) issue(kk + 32, p ^ 1);           // overlap next panel with compute
    if (more) WAIT_ASYNC(5); else WAIT_ASYNC(0);  // in-order: oldest 5 landed
    __syncthreads();

    v16bf a;
    const __bf16* ap = As[p] + (wave * 16 + lm) * LDA_PAD;
    *((int4*)&a)     = *(const int4*)(ap + hb * 8);
    *((int4*)&a + 1) = *(const int4*)(ap + 16 + hb * 8);

    // ping-pong B fragments so ds_load of frag i+1 overlaps WMMA of frag i
    v16bf bfr[2];
    auto loadB = [&](int idx, int which) {     // idx = g*4 + t
      const int g = idx >> 2, t = idx & 3;
      const __bf16* bp = Bs[g][p] + (t * 16 + lm) * LDA_PAD + hb * 8;
      *((int4*)&bfr[which])     = *(const int4*)(bp);
      *((int4*)&bfr[which] + 1) = *(const int4*)(bp + 16);
    };
    loadB(0, 0);
#pragma unroll
    for (int idx = 0; idx < 12; ++idx) {
      if (idx + 1 < 12) loadB(idx + 1, (idx + 1) & 1);
      acc[idx >> 2][idx & 3] =
          __builtin_amdgcn_wmma_f32_16x16x32_bf16(false, a, false, bfr[idx & 1],
                                                  (short)0, acc[idx >> 2][idx & 3],
                                                  false, false);
    }
    __syncthreads();
  }

  const float* bias[3] = {b0, b1, b2};
  float*       outp[3] = {o0, o1, o2};
  const float  scl[3]  = {1.0f, s1scale, 1.0f};
#pragma unroll
  for (int g = 0; g < 3; ++g)
#pragma unroll
    for (int t = 0; t < 4; ++t) {
      const int n = n0 + t * 16 + lm;
      const float bv = bias[g][n];
#pragma unroll
      for (int v = 0; v < 8; ++v) {
        const int m = m0 + wave * 16 + v + 8 * hb;
        outp[g][(size_t)m * N + n] = (acc[g][t][v] + bv) * scl[g];
      }
    }
}

// ---------------------------------------------------------------- bf16 GEMM
// Cout[m,n] = (sum_k A[m,k]*Wt[n,k] + bias[n]) * scale  (double-buffered)
__global__ __launch_bounds__(256) void k_gemm_bf16(const __bf16* __restrict__ A,
    const __bf16* __restrict__ Wt, const float* __restrict__ bias,
    float* __restrict__ Cout, int M, int N, int K, float scale)
{
  __shared__ __bf16 As[2][128 * LDA_PAD];
  __shared__ __bf16 Bs[2][64 * LDA_PAD];
  const int tid  = threadIdx.x;
  const int wave = tid >> 5, lane = tid & 31;
  const int hb = lane >> 4, lm = lane & 15;
  const int m0 = blockIdx.y * 128, n0 = blockIdx.x * 64;

  v8f acc[4];
#pragma unroll
  for (int t = 0; t < 4; ++t) acc[t] = v8f_zero();

  const int arow = tid >> 1, ach = tid & 1;
  const int brow = tid >> 2, bch = tid & 3;

  auto issue = [&](int kk, int p) {
    const __bf16* asrc = A + (size_t)(m0 + arow) * K + kk + ach * 16;
    __bf16*       adst = As[p] + arow * LDA_PAD + ach * 16;
    async_copy_b128(asrc,     adst);
    async_copy_b128(asrc + 8, adst + 8);
    async_copy_b128(Wt + (size_t)(n0 + brow) * K + kk + bch * 8,
                    Bs[p] + brow * LDA_PAD + bch * 8);
  };

  issue(0, 0);
  for (int kk = 0, it = 0; kk < K; kk += 32, ++it) {
    const int p = it & 1;
    const bool more = (kk + 32 < K);
    if (more) issue(kk + 32, p ^ 1);
    if (more) WAIT_ASYNC(3); else WAIT_ASYNC(0);
    __syncthreads();

    v16bf a;
    const __bf16* ap = As[p] + (wave * 16 + lm) * LDA_PAD;
    *((int4*)&a)     = *(const int4*)(ap + hb * 8);
    *((int4*)&a + 1) = *(const int4*)(ap + 16 + hb * 8);

    v16bf bfr[2];
    auto loadB = [&](int t, int which) {
      const __bf16* bp = Bs[p] + (t * 16 + lm) * LDA_PAD + hb * 8;
      *((int4*)&bfr[which])     = *(const int4*)(bp);
      *((int4*)&bfr[which] + 1) = *(const int4*)(bp + 16);
    };
    loadB(0, 0);
#pragma unroll
    for (int t = 0; t < 4; ++t) {
      if (t + 1 < 4) loadB(t + 1, (t + 1) & 1);
      acc[t] = __builtin_amdgcn_wmma_f32_16x16x32_bf16(false, a, false, bfr[t & 1],
                                                       (short)0, acc[t],
                                                       false, false);
    }
    __syncthreads();
  }

#pragma unroll
  for (int t = 0; t < 4; ++t) {
    const int n = n0 + t * 16 + lm;
    const float bv = bias[n];
#pragma unroll
    for (int v = 0; v < 8; ++v) {
      const int m = m0 + wave * 16 + v + 8 * hb;
      Cout[(size_t)m * N + n] = (acc[t][v] + bv) * scale;
    }
  }
}

// ------------------------------------------------- theta-shift + bf16 repack
__global__ __launch_bounds__(256) void k_theta_pack(const float* __restrict__ qf,
    const float* __restrict__ kf, const float* __restrict__ vf,
    const float* __restrict__ sn, const float* __restrict__ cs,
    __bf16* __restrict__ qr, __bf16* __restrict__ kr, __bf16* __restrict__ vr)
{
  const int npairs = DIM_M * (DIM_C / 2);
  int i = blockIdx.x * 256 + threadIdx.x;
  if (i >= npairs) return;
  const int c2 = i % (DIM_C / 2);
  const int ml = i / (DIM_C / 2);           // b*L + l
  const int l  = ml % DIM_L;
  const int b  = ml / DIM_L;
  const int c  = 2 * c2;
  const int nh = c >> 6, d = c & 63;
  const size_t src = (size_t)ml * DIM_C + c;
  const size_t sci = (size_t)l * DIM_HD + d;
  const float s0 = sn[sci], s1 = sn[sci + 1];
  const float c0 = cs[sci], c1 = cs[sci + 1];
  const size_t dst = (((size_t)b * DIM_NH + nh) * DIM_L + l) * DIM_HD + d;

  float q0 = qf[src], q1 = qf[src + 1];
  qr[dst]     = (__bf16)(q0 * c0 - q1 * s0);
  qr[dst + 1] = (__bf16)(q1 * c1 + q0 * s1);
  float k0 = kf[src], k1 = kf[src + 1];
  kr[dst]     = (__bf16)(k0 * c0 - k1 * s0);
  kr[dst + 1] = (__bf16)(k1 * c1 + k0 * s1);
  vr[dst]     = (__bf16)vf[src];
  vr[dst + 1] = (__bf16)vf[src + 1];
}

// ------------------------------------------------------- streaming attention
// grid (L/128, B*NH), 256 threads (8 waves); each wave owns a 16-row Q strip.
__global__ __launch_bounds__(256) void k_attn(const __bf16* __restrict__ qr,
    const __bf16* __restrict__ kr, const __bf16* __restrict__ vr,
    const float* __restrict__ mask, float* __restrict__ attn)
{
  __shared__ __bf16 Plds[8 * 16 * 32];   // per-wave P staging (bf16, row stride 32)
  __shared__ __bf16 Vt[64 * LDA_PAD];    // V tile transposed: [d][key], 64x32
  const int tid  = threadIdx.x;
  const int wave = tid >> 5, lane = tid & 31;
  const int hb = lane >> 4, lm = lane & 15;
  const int bh = blockIdx.y;
  const int b  = bh / DIM_NH, nh = bh % DIM_NH;
  const int m0 = blockIdx.x * 128 + wave * 16;
  const __bf16* Qb = qr + (size_t)bh * DIM_L * DIM_HD;
  const __bf16* Kb = kr + (size_t)bh * DIM_L * DIM_HD;
  const __bf16* Vb = vr + (size_t)bh * DIM_L * DIM_HD;
  const float*  Mb = mask + (size_t)nh * DIM_L * DIM_L;

  // Q fragments kept in registers for the whole key sweep (16x64 -> 2x v16bf)
  v16bf qa[2];
  {
    const __bf16* qrow = Qb + (size_t)(m0 + lm) * DIM_HD;
#pragma unroll
    for (int f = 0; f < 2; ++f) {
      *((int4*)&qa[f])     = *(const int4*)(qrow + f * 32 + hb * 8);
      *((int4*)&qa[f] + 1) = *(const int4*)(qrow + f * 32 + 16 + hb * 8);
    }
  }

  v8f o[4];
#pragma unroll
  for (int t = 0; t < 4; ++t) o[t] = v8f_zero();
  float rmax[8], rsum[8];
#pragma unroll
  for (int v = 0; v < 8; ++v) { rmax[v] = -3.0e38f; rsum[v] = 0.0f; }

  const int vrow = tid >> 3, vch = tid & 7;   // V loader: 32 rows x 8 chunks of 8

  for (int j0 = 0; j0 < DIM_L; j0 += 32) {
    __syncthreads();   // previous iteration's Vt reads complete
    { // stage V tile transposed: Vt[d][key]; b128 gather then LDS scatter
      __bf16 tmp[8];
      *(int4*)&tmp[0] = *(const int4*)(Vb + (size_t)(j0 + vrow) * DIM_HD + vch * 8);
#pragma unroll
      for (int e = 0; e < 8; ++e)
        Vt[(vch * 8 + e) * LDA_PAD + vrow] = tmp[e];
    }
    __syncthreads();

    // --- S = Q K^T + mask, two 16x16 column tiles; ping-pong K fragments ---
    v8f sc[2];
    sc[0] = v8f_zero(); sc[1] = v8f_zero();
    {
      v16bf kb[2];
      auto loadK = [&](int idx, int which) {   // idx = cc*2 + f
        const int cc = idx >> 1, f = idx & 1;
        const __bf16* krow =
            Kb + (size_t)(j0 + cc * 16 + lm) * DIM_HD + f * 32 + hb * 8;
        *((int4*)&kb[which])     = *(const int4*)(krow);
        *((int4*)&kb[which] + 1) = *(const int4*)(krow + 16);
      };
      loadK(0, 0);
#pragma unroll
      for (int idx = 0; idx < 4; ++idx) {
        if (idx + 1 < 4) loadK(idx + 1, (idx + 1) & 1);
        sc[idx >> 1] = __builtin_amdgcn_wmma_f32_16x16x32_bf16(
            false, qa[idx & 1], false, kb[idx & 1], (short)0, sc[idx >> 1],
            false, false);
      }
    }
#pragma unroll
    for (int cc = 0; cc < 2; ++cc)
#pragma unroll
      for (int v = 0; v < 8; ++v)
        sc[cc][v] += Mb[(size_t)(m0 + v + 8 * hb) * DIM_L + (j0 + cc * 16 + lm)];

    // --- online softmax: row lives in one VGPR across 16 lanes of a half ---
#pragma unroll
    for (int v = 0; v < 8; ++v) {
      float mv = fmaxf(sc[0][v], sc[1][v]);
      mv = fmaxf(mv, __shfl_xor(mv, 1, 32));
      mv = fmaxf(mv, __shfl_xor(mv, 2, 32));
      mv = fmaxf(mv, __shfl_xor(mv, 4, 32));
      mv = fmaxf(mv, __shfl_xor(mv, 8, 32));
      const float nm   = fmaxf(rmax[v], mv);
      const float corr = __expf(rmax[v] - nm);
      rmax[v] = nm;
      const float p0 = __expf(sc[0][v] - nm);
      const float p1 = __expf(sc[1][v] - nm);
      sc[0][v] = p0; sc[1][v] = p1;
      float ls = p0 + p1;
      ls += __shfl_xor(ls, 1, 32);
      ls += __shfl_xor(ls, 2, 32);
      ls += __shfl_xor(ls, 4, 32);
      ls += __shfl_xor(ls, 8, 32);
      rsum[v] = rsum[v] * corr + ls;
#pragma unroll
      for (int t = 0; t < 4; ++t) o[t][v] *= corr;
    }

    // --- O += P V : stage P (C layout) -> LDS -> re-read as A operand ---
    __bf16* Pw = Plds + wave * 512;
#pragma unroll
    for (int cc = 0; cc < 2; ++cc)
#pragma unroll
      for (int v = 0; v < 8; ++v)
        Pw[(v + 8 * hb) * 32 + cc * 16 + lm] = (__bf16)sc[cc][v];

    v16bf pa;   // A-operand: 16 rows x 32 keys
    {
      const __bf16* prow = Pw + lm * 32;
      *((int4*)&pa)     = *(const int4*)(prow + hb * 8);
      *((int4*)&pa + 1) = *(const int4*)(prow + 16 + hb * 8);
    }
    {
      v16bf vb[2];
      auto loadV = [&](int t, int which) {
        const __bf16* vtrow = Vt + (t * 16 + lm) * LDA_PAD + hb * 8;
        *((int4*)&vb[which])     = *(const int4*)(vtrow);
        *((int4*)&vb[which] + 1) = *(const int4*)(vtrow + 16);
      };
      loadV(0, 0);
#pragma unroll
      for (int t = 0; t < 4; ++t) {
        if (t + 1 < 4) loadV(t + 1, (t + 1) & 1);
        o[t] = __builtin_amdgcn_wmma_f32_16x16x32_bf16(false, pa, false, vb[t & 1],
                                                       (short)0, o[t],
                                                       false, false);
      }
    }
  }

  // write normalized output straight into [B, L, C] layout (c = nh*64 + d)
#pragma unroll
  for (int t = 0; t < 4; ++t) {
    const int d = t * 16 + lm;
#pragma unroll
    for (int v = 0; v < 8; ++v) {
      const int l = m0 + v + 8 * hb;
      attn[((size_t)b * DIM_L + l) * DIM_C + nh * DIM_HD + d] = o[t][v] / rsum[v];
    }
  }
}

// -------------------------------------- depthwise 5x5 lepe + add, emit bf16
__global__ __launch_bounds__(256) void k_lepe_add(const float* __restrict__ attn,
    const float* __restrict__ vf, const float* __restrict__ lw,
    const float* __restrict__ lb, __bf16* __restrict__ fused)
{
  int i = blockIdx.x * 256 + threadIdx.x;
  if (i >= DIM_M * DIM_C) return;
  const int c  = i % DIM_C;
  const int ml = i / DIM_C;
  const int l  = ml % DIM_L;
  const int b  = ml / DIM_L;
  const int h = l >> 5, w = l & 31;
  float acc = attn[i] + lb[c];
#pragma unroll
  for (int dh = 0; dh < 5; ++dh) {
    const int hh = h + dh - 2;
    if (hh < 0 || hh >= DIM_H) continue;
#pragma unroll
    for (int dw = 0; dw < 5; ++dw) {
      const int ww = w + dw - 2;
      if (ww < 0 || ww >= DIM_W) continue;
      acc += vf[((size_t)b * DIM_L + hh * DIM_W + ww) * DIM_C + c] *
             lw[(dh * 5 + dw) * DIM_C + c];
    }
  }
  fused[i] = (__bf16)acc;
}

// ---------------------------------------------------------------------------
extern "C" void kernel_launch(void* const* d_in, const int* in_sizes, int n_in,
                              void* d_out, int out_size, void* d_ws, size_t ws_size,
                              hipStream_t stream) {
  (void)in_sizes; (void)n_in; (void)out_size; (void)ws_size;
  const float* x     = (const float*)d_in[0];
  const float* sinp  = (const float*)d_in[1];
  const float* cosp  = (const float*)d_in[2];
  const float* mask  = (const float*)d_in[3];
  const float* wq    = (const float*)d_in[4];
  const float* bq    = (const float*)d_in[5];
  const float* wk    = (const float*)d_in[6];
  const float* bk    = (const float*)d_in[7];
  const float* wv    = (const float*)d_in[8];
  const float* bv    = (const float*)d_in[9];
  const float* lepew = (const float*)d_in[10];
  const float* lepeb = (const float*)d_in[11];
  const float* wo    = (const float*)d_in[12];
  const float* bo    = (const float*)d_in[13];

  char* wsp = (char*)d_ws;
  auto carve = [&](size_t bytes) {
    void* p = (void*)wsp;
    wsp += (bytes + 255) & ~(size_t)255;
    return p;
  };
  const size_t MC = (size_t)DIM_M * DIM_C;
  __bf16* xbf  = (__bf16*)carve(MC * 2);
  __bf16* wqbf = (__bf16*)carve((size_t)DIM_C * DIM_C * 2);
  __bf16* wkbf = (__bf16*)carve((size_t)DIM_C * DIM_C * 2);
  __bf16* wvbf = (__bf16*)carve((size_t)DIM_C * DIM_C * 2);
  __bf16* wobf = (__bf16*)carve((size_t)DIM_C * DIM_C * 2);
  float*  qf   = (float*)carve(MC * 4);
  float*  kf   = (float*)carve(MC * 4);
  float*  vfb  = (float*)carve(MC * 4);
  __bf16* qrbf = (__bf16*)carve(MC * 2);
  __bf16* krbf = (__bf16*)carve(MC * 2);
  __bf16* vrbf = (__bf16*)carve(MC * 2);
  float*  attnf  = qf;              // reuse: qf dead after theta_pack
  __bf16* fusedb = (__bf16*)kf;     // reuse: kf dead after theta_pack

  const int WC = DIM_C * DIM_C;     // 589824
  k_cvt_bf16<<<(int)(MC / 1024), 256, 0, stream>>>(x, xbf, (int)MC);
  k_cvt_bf16<<<WC / 1024, 256, 0, stream>>>(wq, wqbf, WC);
  k_cvt_bf16<<<WC / 1024, 256, 0, stream>>>(wk, wkbf, WC);
  k_cvt_bf16<<<WC / 1024, 256, 0, stream>>>(wv, wvbf, WC);
  k_cvt_bf16<<<WC / 1024, 256, 0, stream>>>(wo, wobf, WC);

  const dim3 gemm_grid(DIM_C / 64, DIM_M / 128);   // (12, 64)
  const float kscale = 0.125f;                     // (C/NH)^-0.5 = 64^-0.5

  // fused Q/K/V projection: X tile staged once, 12 WMMAs/wave/K-panel
  k_gemm_qkv<<<gemm_grid, 256, 0, stream>>>(xbf, wqbf, wkbf, wvbf,
                                            bq, bk, bv, qf, kf, vfb, kscale);

  k_theta_pack<<<(DIM_M * (DIM_C / 2)) / 256, 256, 0, stream>>>(
      qf, kf, vfb, sinp, cosp, qrbf, krbf, vrbf);

  k_attn<<<dim3(DIM_L / 128, DIM_B * DIM_NH), 256, 0, stream>>>(
      qrbf, krbf, vrbf, mask, attnf);

  k_lepe_add<<<(int)(MC / 256), 256, 0, stream>>>(attnf, vfb, lepew, lepeb, fusedb);

  k_gemm_bf16<<<gemm_grid, 256, 0, stream>>>(fusedb, wobf, bo, (float*)d_out,
                                             DIM_M, DIM_C, DIM_C, 1.0f);
}
